// Attention_85907935855327
// MI455X (gfx1250) — matrix operand order repaired and checked by reading, not compiled, not run on hardware
//
#include <hip/hip_runtime.h>
#include <hip/hip_bf16.h>

typedef __bf16 bf16_t;
typedef __attribute__((ext_vector_type(16))) __bf16 v16bf;
typedef __attribute__((ext_vector_type(8)))  __bf16 v8bf;
typedef __attribute__((ext_vector_type(8)))  float  v8f;

union ABFrag { v16bf v; v8bf h[2]; };

#define WMMA_BF16(A, Bm, C) \
  __builtin_amdgcn_wmma_f32_16x16x32_bf16(false, (A), false, (Bm), (short)0, (C), false, false)

// Problem constants
#define BB    128
#define SS    196
#define DIN   512
#define HH    8
#define DKK   32
#define DVV   128
#define QKVD  1536
#define PIN   1024
#define MROWS (BB*SS)      // 25088
#define SPAD  208          // S padded to 13*16 for key tiles
#define SPAD2 224          // S padded to 7*32 for AV k-steps
#define ATT_SCALE 0.17677669529663687f  // 32^-0.5

// ---- CDNA5 async copy (global -> LDS), tracked by ASYNCcnt -----------------
__device__ __forceinline__ void async_ld16(unsigned lds_off, const void* gptr) {
  asm volatile("global_load_async_to_lds_b128 %0, %1, off"
               :: "v"(lds_off), "v"((unsigned long long)(uintptr_t)gptr)
               : "memory");
}
__device__ __forceinline__ void wait_async0() {
  asm volatile("s_wait_asynccnt 0x0" ::: "memory");
}

// ---------------------------------------------------------------- prep kernels

__global__ void prep_x_bf16(const float* __restrict__ x, bf16_t* __restrict__ xb, long n) {
  long i = (long)blockIdx.x * blockDim.x + threadIdx.x;
  long stride = (long)gridDim.x * blockDim.x;
  for (; i < n; i += stride) xb[i] = (bf16_t)x[i];
}

// W_qkv [512,1536] -> WqT [1536,512] bf16 with BN1 scale folded; bias1[n] f32
__global__ void prep_wq(const float* __restrict__ W, const float* __restrict__ bq,
                        const float* __restrict__ g, const float* __restrict__ be,
                        const float* __restrict__ mu, const float* __restrict__ var,
                        bf16_t* __restrict__ wqt, float* __restrict__ bias1) {
  int i = blockIdx.x * blockDim.x + threadIdx.x;
  if (i >= QKVD * DIN) return;
  int n = i / DIN, kk = i - n * DIN;
  float a = g[n] * rsqrtf(var[n] + 1e-3f);
  wqt[(size_t)n * DIN + kk] = (bf16_t)(W[(size_t)kk * QKVD + n] * a);
  if (kk == 0) bias1[n] = (bq[n] - mu[n]) * a + be[n];
}

// W_proj [1024,512] -> WpT [512,1024] bf16 with BN2 scale folded; bias2[n] f32
__global__ void prep_wp(const float* __restrict__ W, const float* __restrict__ bp,
                        const float* __restrict__ g, const float* __restrict__ be,
                        const float* __restrict__ mu, const float* __restrict__ var,
                        bf16_t* __restrict__ wpt, float* __restrict__ bias2) {
  int i = blockIdx.x * blockDim.x + threadIdx.x;
  if (i >= DIN * PIN) return;
  int n = i / PIN, kk = i - n * PIN;
  float a = g[n] * rsqrtf(var[n] + 1e-3f);
  wpt[(size_t)n * PIN + kk] = (bf16_t)(W[(size_t)kk * DIN + n] * a);
  if (kk == 0) bias2[n] = (bp[n] - mu[n]) * a + be[n];
}

__global__ void zero_u32(unsigned int* __restrict__ p, long n) {
  long i = (long)blockIdx.x * blockDim.x + threadIdx.x;
  long stride = (long)gridDim.x * blockDim.x;
  for (; i < n; i += stride) p[i] = 0u;
}

// ------------------------------------------------- GEMM1: qkv = x @ WqT + b1
// Block: 256 thr = 8 waves; block tile 128(M) x 128(N); wave tile 64x32.
// A/B 128x32 bf16 tiles staged in LDS via async copy, double-buffered.
__global__ __launch_bounds__(256)
void gemm_qkv(const bf16_t* __restrict__ xb, const bf16_t* __restrict__ wqt,
              const float* __restrict__ bias1,
              bf16_t* __restrict__ qw, bf16_t* __restrict__ kw,
              bf16_t* __restrict__ vtw) {
  __shared__ bf16_t sA[2][128 * 32];
  __shared__ bf16_t sB[2][128 * 32];

  const int tid  = threadIdx.x;
  const int lane = tid & 31;
  const int wv   = tid >> 5;
  const int bm0 = blockIdx.x * 128;
  const int bn0 = blockIdx.y * 128;
  const int wm = (wv & 1) * 64;       // wave M offset within block
  const int wn = (wv >> 1) * 32;      // wave N offset within block
  const int ml = lane & 15;
  const int hi = lane >> 4;
  const int ak0 = hi * 8;             // A-frag K sub-offset
  const int bk0 = hi * 8;             // B-frag K sub-offset

  v8f z = {0.f,0.f,0.f,0.f,0.f,0.f,0.f,0.f};
  v8f acc[4][2];
  for (int mt = 0; mt < 4; ++mt) for (int nt = 0; nt < 2; ++nt) acc[mt][nt] = z;

  const char* gA = (const char*)(xb  + (size_t)bm0 * DIN);
  const char* gB = (const char*)(wqt + (size_t)bn0 * DIN);
  const unsigned baseA[2] = { (unsigned)(uintptr_t)&sA[0][0], (unsigned)(uintptr_t)&sA[1][0] };
  const unsigned baseB[2] = { (unsigned)(uintptr_t)&sB[0][0], (unsigned)(uintptr_t)&sB[1][0] };

  // tile = 128 rows x 64B = 512 chunks of 16B; chunk c: row=c>>2, byte=(c&3)*16.
  // Each thread copies chunks {tid, tid+256} of A and B — straight-line, no loop.
  const int c0 = tid, c1 = tid + 256;
  const size_t gc0 = (size_t)(c0 >> 2) * (DIN * 2) + (size_t)((c0 & 3) * 16);
  const size_t gc1 = (size_t)(c1 >> 2) * (DIN * 2) + (size_t)((c1 & 3) * 16);
  auto issueTile = [&](int buf, int kk) {
    size_t kb = (size_t)kk * 2;
    async_ld16(baseA[buf] + c0 * 16, gA + gc0 + kb);
    async_ld16(baseA[buf] + c1 * 16, gA + gc1 + kb);
    async_ld16(baseB[buf] + c0 * 16, gB + gc0 + kb);
    async_ld16(baseB[buf] + c1 * 16, gB + gc1 + kb);
  };

  issueTile(0, 0);
  wait_async0();
  __syncthreads();

  const int NK = DIN / 32;   // 16
  for (int i = 0; i < NK; ++i) {
    int cur = i & 1;
    if (i + 1 < NK) issueTile(cur ^ 1, (i + 1) * 32);

    const bf16_t* A  = sA[cur];
    const bf16_t* Bm = sB[cur];
    ABFrag a[4], b[2];
    for (int mt = 0; mt < 4; ++mt) {
      const bf16_t* r = A + (wm + mt*16 + ml) * 32 + ak0;
      a[mt].h[0] = *(const v8bf*)(r);
      a[mt].h[1] = *(const v8bf*)(r + 16);
    }
    for (int nt = 0; nt < 2; ++nt) {
      const bf16_t* r = Bm + (wn + nt*16 + ml) * 32 + bk0;
      b[nt].h[0] = *(const v8bf*)(r);
      b[nt].h[1] = *(const v8bf*)(r + 16);
    }
    for (int mt = 0; mt < 4; ++mt)
      for (int nt = 0; nt < 2; ++nt)
        acc[mt][nt] = WMMA_BF16(a[mt].v, b[nt].v, acc[mt][nt]);

    wait_async0();
    __syncthreads();
  }

  // Epilogue: add bias, scatter q / k row-major (S padded), v transposed [DV][SPAD2]
  for (int mt = 0; mt < 4; ++mt)
    for (int nt = 0; nt < 2; ++nt)
      for (int r = 0; r < 8; ++r) {
        int M = bm0 + wm + mt*16 + r + hi*8;  // global row, < 25088
        int C = bn0 + wn + nt*16 + ml;        // global col, < 1536
        float v = acc[mt][nt][r] + bias1[C];
        int b_  = M / SS;
        int s   = M - b_ * SS;
        int h_  = C / (2*DKK + DVV);          // /192
        int j   = C - h_ * (2*DKK + DVV);
        int bh  = b_ * HH + h_;
        bf16_t o = (bf16_t)v;
        if (j < DKK)          qw[((size_t)bh * SPAD + s) * DKK + j]          = o;
        else if (j < 2*DKK)   kw[((size_t)bh * SPAD + s) * DKK + (j - DKK)]  = o;
        else                  vtw[((size_t)bh * DVV + (j - 2*DKK)) * SPAD2 + s] = o;
      }
}

// ------------------------------------------------- attention per (b,h)
// Block: 128 thr = 4 waves; wave handles query tiles wv, wv+4, wv+8, wv+12 (<13).
__global__ __launch_bounds__(128)
void attn(const bf16_t* __restrict__ qw, const bf16_t* __restrict__ kw,
          const bf16_t* __restrict__ vtw, bf16_t* __restrict__ hid) {
  __shared__ bf16_t pl[4][16][SPAD2];

  const int lane = threadIdx.x & 31;
  const int wv   = threadIdx.x >> 5;
  const int bh   = blockIdx.x;         // 0..1023
  const int b_   = bh >> 3, h_ = bh & 7;
  const int ml   = lane & 15;
  const int hi   = lane >> 4;

  const bf16_t* q  = qw  + (size_t)bh * SPAD * DKK;
  const bf16_t* k  = kw  + (size_t)bh * SPAD * DKK;
  const bf16_t* vt = vtw + (size_t)bh * DVV * SPAD2;
  bf16_t* hout = hid + (size_t)b_ * (SS * PIN) + (size_t)h_ * (DVV * SS);

  // zero the 208..223 pad columns of this wave's P buffer once
  for (int i = lane; i < 16 * 16; i += 32)
    pl[wv][i >> 4][SPAD + (i & 15)] = (bf16_t)0.f;

  for (int qt = wv; qt < 13; qt += 4) {
    // Q A-fragment (DK=32 => exactly one WMMA k-step)
    ABFrag qa;
    {
      const bf16_t* r = q + (size_t)(qt*16 + ml) * DKK + hi*8;
      qa.h[0] = *(const v8bf*)(r);
      qa.h[1] = *(const v8bf*)(r + 16);
    }

    // scores = Q K^T
    v8f sc[13];
    for (int kt = 0; kt < 13; ++kt) {
      ABFrag kb;
      const bf16_t* r = k + (size_t)(kt*16 + ml) * DKK + hi*8;
      kb.h[0] = *(const v8bf*)(r);
      kb.h[1] = *(const v8bf*)(r + 16);
      v8f zz = {0.f,0.f,0.f,0.f,0.f,0.f,0.f,0.f};
      sc[kt] = WMMA_BF16(qa.v, kb.v, zz);
    }
    // scale + mask padded keys
    for (int kt = 0; kt < 13; ++kt) {
      int n = kt*16 + ml;
      for (int r = 0; r < 8; ++r) {
        float x = sc[kt][r] * ATT_SCALE;
        sc[kt][r] = (n < SS) ? x : -1e30f;
      }
    }
    // row-wise softmax (rows striped: VGPR r, half-wave holds 16 cols)
    float inv[8];
    for (int r = 0; r < 8; ++r) {
      float mx = -1e30f;
      for (int kt = 0; kt < 13; ++kt) mx = fmaxf(mx, sc[kt][r]);
      for (int m = 1; m < 16; m <<= 1) mx = fmaxf(mx, __shfl_xor(mx, m, 32));
      float sum = 0.f;
      for (int kt = 0; kt < 13; ++kt) {
        float e = __expf(sc[kt][r] - mx);
        sc[kt][r] = e; sum += e;
      }
      for (int m = 1; m < 16; m <<= 1) sum += __shfl_xor(sum, m, 32);
      inv[r] = 1.f / sum;
    }
    // stage P in LDS (C-layout -> row-major) for re-read in A-layout
    for (int kt = 0; kt < 13; ++kt) {
      int n = kt*16 + ml;
      for (int r = 0; r < 8; ++r)
        pl[wv][r + hi*8][n] = (bf16_t)(sc[kt][r] * inv[r]);
    }

    // AV = P (16 x 224) x V^T (224 x 128), 7 k-steps, 8 n-tiles
    for (int nt = 0; nt < 8; ++nt) {
      v8f acc = {0.f,0.f,0.f,0.f,0.f,0.f,0.f,0.f};
      for (int kt = 0; kt < 7; ++kt) {
        ABFrag pa, vb;
        const bf16_t* pr = &pl[wv][ml][kt*32 + hi*8];
        pa.h[0] = *(const v8bf*)(pr);
        pa.h[1] = *(const v8bf*)(pr + 16);
        const bf16_t* vr = vt + (size_t)(nt*16 + ml) * SPAD2 + kt*32 + hi*8;
        vb.h[0] = *(const v8bf*)(vr);
        vb.h[1] = *(const v8bf*)(vr + 16);
        acc = WMMA_BF16(pa.v, vb.v, acc);
      }
      // hard_swish + faithful [B,H,DV,S] -> [B,S,H*DV] scatter
      for (int r = 0; r < 8; ++r) {
        int s = qt*16 + r + hi*8;
        if (s < SS) {
          int d = nt*16 + ml;
          float x = acc[r];
          float y = x * fminf(fmaxf(x + 3.f, 0.f), 6.f) * (1.f/6.f);
          hout[(size_t)d * SS + s] = (bf16_t)y;
        }
      }
    }
  }
}

// ------------------------------------------------- GEMM2: out = hidden @ WpT + b2
__global__ __launch_bounds__(256)
void gemm_proj(const bf16_t* __restrict__ hid, const bf16_t* __restrict__ wpt,
               const float* __restrict__ bias2, float* __restrict__ out) {
  __shared__ bf16_t sA[2][128 * 32];
  __shared__ bf16_t sB[2][128 * 32];

  const int tid  = threadIdx.x;
  const int lane = tid & 31;
  const int wv   = tid >> 5;
  const int bm0 = blockIdx.x * 128;
  const int bn0 = blockIdx.y * 128;
  const int wm = (wv & 1) * 64;
  const int wn = (wv >> 1) * 32;
  const int ml = lane & 15;
  const int hi = lane >> 4;
  const int ak0 = hi * 8;
  const int bk0 = hi * 8;

  v8f z = {0.f,0.f,0.f,0.f,0.f,0.f,0.f,0.f};
  v8f acc[4][2];
  for (int mt = 0; mt < 4; ++mt) for (int nt = 0; nt < 2; ++nt) acc[mt][nt] = z;

  const char* gA = (const char*)(hid + (size_t)bm0 * PIN);
  const char* gB = (const char*)(wpt + (size_t)bn0 * PIN);
  const unsigned baseA[2] = { (unsigned)(uintptr_t)&sA[0][0], (unsigned)(uintptr_t)&sA[1][0] };
  const unsigned baseB[2] = { (unsigned)(uintptr_t)&sB[0][0], (unsigned)(uintptr_t)&sB[1][0] };

  const int c0 = tid, c1 = tid + 256;
  const size_t gc0 = (size_t)(c0 >> 2) * (PIN * 2) + (size_t)((c0 & 3) * 16);
  const size_t gc1 = (size_t)(c1 >> 2) * (PIN * 2) + (size_t)((c1 & 3) * 16);
  auto issueTile = [&](int buf, int kk) {
    size_t kb = (size_t)kk * 2;
    async_ld16(baseA[buf] + c0 * 16, gA + gc0 + kb);
    async_ld16(baseA[buf] + c1 * 16, gA + gc1 + kb);
    async_ld16(baseB[buf] + c0 * 16, gB + gc0 + kb);
    async_ld16(baseB[buf] + c1 * 16, gB + gc1 + kb);
  };

  issueTile(0, 0);
  wait_async0();
  __syncthreads();

  const int NK = PIN / 32;  // 32
  for (int i = 0; i < NK; ++i) {
    int cur = i & 1;
    if (i + 1 < NK) issueTile(cur ^ 1, (i + 1) * 32);

    const bf16_t* A  = sA[cur];
    const bf16_t* Bm = sB[cur];
    ABFrag a[4], b[2];
    for (int mt = 0; mt < 4; ++mt) {
      const bf16_t* r = A + (wm + mt*16 + ml) * 32 + ak0;
      a[mt].h[0] = *(const v8bf*)(r);
      a[mt].h[1] = *(const v8bf*)(r + 16);
    }
    for (int nt = 0; nt < 2; ++nt) {
      const bf16_t* r = Bm + (wn + nt*16 + ml) * 32 + bk0;
      b[nt].h[0] = *(const v8bf*)(r);
      b[nt].h[1] = *(const v8bf*)(r + 16);
    }
    for (int mt = 0; mt < 4; ++mt)
      for (int nt = 0; nt < 2; ++nt)
        acc[mt][nt] = WMMA_BF16(a[mt].v, b[nt].v, acc[mt][nt]);

    wait_async0();
    __syncthreads();
  }

  for (int mt = 0; mt < 4; ++mt)
    for (int nt = 0; nt < 2; ++nt)
      for (int r = 0; r < 8; ++r) {
        int M = bm0 + wm + mt*16 + r + hi*8;
        int C = bn0 + wn + nt*16 + ml;
        out[(size_t)M * DIN + C] = acc[mt][nt][r] + bias2[C];
      }
}

// ------------------------------------------------- host launch

extern "C" void kernel_launch(void* const* d_in, const int* in_sizes, int n_in,
                              void* d_out, int out_size, void* d_ws, size_t ws_size,
                              hipStream_t stream) {
  (void)in_sizes; (void)n_in; (void)out_size; (void)ws_size;
  const float* x    = (const float*)d_in[0];
  const float* Wq   = (const float*)d_in[1];
  const float* bq   = (const float*)d_in[2];
  const float* g1   = (const float*)d_in[3];
  const float* be1  = (const float*)d_in[4];
  const float* mu1  = (const float*)d_in[5];
  const float* va1  = (const float*)d_in[6];
  const float* Wp   = (const float*)d_in[7];
  const float* bp   = (const float*)d_in[8];
  const float* g2   = (const float*)d_in[9];
  const float* be2  = (const float*)d_in[10];
  const float* mu2  = (const float*)d_in[11];
  const float* va2  = (const float*)d_in[12];
  float* out = (float*)d_out;

  char* ws = (char*)d_ws;
  size_t off = 0;
  auto carve = [&](size_t bytes) -> char* {
    char* p = ws + off;
    off += (bytes + 255) & ~(size_t)255;
    return p;
  };
  bf16_t* xb  = (bf16_t*)carve((size_t)MROWS * DIN * 2);
  bf16_t* wqt = (bf16_t*)carve((size_t)QKVD * DIN * 2);
  float*  b1  = (float*) carve((size_t)QKVD * 4);
  bf16_t* wpt = (bf16_t*)carve((size_t)DIN * PIN * 2);
  float*  b2  = (float*) carve((size_t)DIN * 4);
  bf16_t* qw  = (bf16_t*)carve((size_t)BB*HH * SPAD * DKK * 2);
  bf16_t* kw  = (bf16_t*)carve((size_t)BB*HH * SPAD * DKK * 2);
  bf16_t* vtw = (bf16_t*)carve((size_t)BB*HH * DVV * SPAD2 * 2);
  bf16_t* hid = (bf16_t*)carve((size_t)MROWS * PIN * 2);

  // 1) conversions / BN folding
  long nx = (long)MROWS * DIN;
  prep_x_bf16<<<2048, 256, 0, stream>>>(x, xb, nx);
  prep_wq<<<(QKVD*DIN + 255)/256, 256, 0, stream>>>(Wq, bq, g1, be1, mu1, va1, wqt, b1);
  prep_wp<<<(DIN*PIN + 255)/256, 256, 0, stream>>>(Wp, bp, g2, be2, mu2, va2, wpt, b2);

  // 2) zero q/k/vt pads (regions are contiguous, sizes multiples of 256B)
  long zero_words = ((long)BB*HH*SPAD*DKK*2 * 2 + (long)BB*HH*DVV*SPAD2*2) / 4;
  zero_u32<<<2048, 256, 0, stream>>>((unsigned int*)qw, zero_words);

  // 3) QKV GEMM + BN + scatter
  gemm_qkv<<<dim3(MROWS/128, QKVD/128), 256, 0, stream>>>(xb, wqt, b1, qw, kw, vtw);

  // 4) attention + hard_swish + reshape
  attn<<<BB*HH, 128, 0, stream>>>(qw, kw, vtw, hid);

  // 5) projection GEMM + BN
  gemm_proj<<<dim3(MROWS/128, DIN/128), 256, 0, stream>>>(hid, wpt, b2, out);
}
